// SVDNet_19567871001029
// MI455X (gfx1250) — compile-verified, matrix operand order checked by reading, not run on hardware
//
#include <hip/hip_runtime.h>
#include <hip/hip_bf16.h>
#include <math.h>

typedef __attribute__((ext_vector_type(16))) _Float16 v16h;
typedef __attribute__((ext_vector_type(8)))  _Float16 h8;
typedef __attribute__((ext_vector_type(8)))  float    v8f;

// ---------------- workspace layout (bytes) ----------------
// h1 is zero-PADDED (halo 2): 260 x 260 x 32 f16, HWC -> conv2 loop is branch-free.
#define WS_H1_OFF      0u            // 260*260*32 f16 = 4,326,400 B
#define WS_W2P_OFF     4326400u      // pre-swizzled conv2 weights: 25*4*32*16 f16 = 100 KB
#define WS_CHAN_OFF    4428800u      // 64 f32 channel sums
#define WS_FVEC_OFF    4429056u      // 256 f32 feature vector
#define WS_UBUF_OFF    4430080u      // U staging, column-major (128 cols x 512 rows x 2) f32
#define WS_VBUF_OFF    4954368u      // V staging, same

#define H1W 260                       // padded width/height
// padded pixel (y,x) with y,x in [0,260); interior pixel (oy,ox) at (oy+2, ox+2)

// =====================================================================
// Zero the 2-wide halo border of the padded activation (2064 pixels x 64B).
// =====================================================================
__global__ void __launch_bounds__(256)
k_zero_pad(_Float16* __restrict__ h1p) {
  int t = blockIdx.x * 256 + threadIdx.x;
  if (t >= 2064) return;
  int y, x;
  if (t < 520) {                      // rows 0,1
    y = t / 260; x = t % 260;
  } else if (t < 1040) {              // rows 258,259
    int u = t - 520; y = 258 + u / 260; x = u % 260;
  } else {                            // rows 2..257, cols {0,1,258,259}
    int u = t - 1040;                 // 0..1023
    y = 2 + (u >> 2);
    int c = u & 3;
    x = (c < 2) ? c : (256 + c);      // 0,1,258,259
  }
  float4 z = {0.f, 0.f, 0.f, 0.f};
  float4* dst = (float4*)(h1p + (size_t)(y * H1W + x) * 32);
#pragma unroll
  for (int q = 0; q < 4; ++q) dst[q] = z;
}

// =====================================================================
// conv1: 2->32 ch, 5x5, stride 2, pad 2, +bias, ReLU. Writes interior of
// padded HWC f16 buffer so conv2's WMMA A-fragments are contiguous,
// unconditional 16-byte loads.
// =====================================================================
__global__ void __launch_bounds__(128)
k_conv1(const float* __restrict__ x, const float* __restrict__ w,
        const float* __restrict__ b, _Float16* __restrict__ h1p) {
  int pix = blockIdx.x * 128 + threadIdx.x;      // 0..65535
  int oy = pix >> 8, ox = pix & 255;
  float patch[50];
#pragma unroll
  for (int ky = 0; ky < 5; ++ky) {
    int iy = 2 * oy - 2 + ky;
    bool yok = (iy >= 0) && (iy < 512);
#pragma unroll
    for (int kx = 0; kx < 5; ++kx) {
      int ix = 2 * ox - 2 + kx;
      bool ok = yok && (ix >= 0) && (ix < 512);
      const float* p = x + ((iy * 512 + ix) << 1);
      patch[(0 * 5 + ky) * 5 + kx] = ok ? p[0] : 0.f;
      patch[(1 * 5 + ky) * 5 + kx] = ok ? p[1] : 0.f;
    }
  }
  union { _Float16 h[32]; float4 f4[4]; } out;
#pragma unroll 4
  for (int oc = 0; oc < 32; ++oc) {
    float acc = b[oc];
    const float* wp = w + oc * 50;
#pragma unroll
    for (int q = 0; q < 50; ++q) acc += wp[q] * patch[q];
    out.h[oc] = (_Float16)fmaxf(acc, 0.f);
  }
  float4* dst = (float4*)(h1p + (size_t)((oy + 2) * H1W + (ox + 2)) * 32);
#pragma unroll
  for (int q = 0; q < 4; ++q) dst[q] = out.f4[q];
}

// =====================================================================
// Pre-swizzle conv2 weights into per-lane WMMA B-fragment order:
// layout [tap(25)][oc_tile(4)][lane(32)][16 f16]. Also zero chansum.
// B(k=ic, n=oc): lane holds n=lane%16, k runs [half*8,+8) and [16+half*8,+8).
// =====================================================================
__global__ void __launch_bounds__(128)
k_pack_w2(const float* __restrict__ w2, _Float16* __restrict__ w2p,
          float* __restrict__ chansum) {
  int tap = blockIdx.x;                 // 0..24
  int t = threadIdx.x;                  // 0..127
  int oct = t >> 5, lane = t & 31;
  int n = lane & 15, half = lane >> 4;
  int oc = oct * 16 + n;
  _Float16 frag[16];
#pragma unroll
  for (int e = 0; e < 8; ++e) {
    int ic0 = half * 8 + e;
    int ic1 = 16 + half * 8 + e;
    frag[e]     = (_Float16)w2[(oc * 32 + ic0) * 25 + tap];
    frag[e + 8] = (_Float16)w2[(oc * 32 + ic1) * 25 + tap];
  }
  _Float16* dst = w2p + (size_t)(((tap * 4 + oct) * 32 + lane) * 16);
#pragma unroll
  for (int e = 0; e < 16; ++e) dst[e] = frag[e];
  if (blockIdx.x == 0 && t < 64) chansum[t] = 0.f;
}

// =====================================================================
// conv2 as implicit GEMM with v_wmma_f32_16x16x32_f16, branch-free.
// Wave = 16 output pixels x 16 out-channels; K = 800 = 25 taps x 32 ic.
// Padded input -> no bounds checks, no exec masking in the WMMA loop.
// Epilogue: ReLU + pixel reduction + atomicAdd (fused global mean pool).
// =====================================================================
__global__ void __launch_bounds__(128)
k_conv2_wmma(const _Float16* __restrict__ h1p, const _Float16* __restrict__ w2p,
             const float* __restrict__ b2, float* __restrict__ chansum) {
  int lane = threadIdx.x & 31;
  int oct  = threadIdx.x >> 5;          // oc tile 0..3
  int oxt  = blockIdx.x;                // 0..7
  int oy   = blockIdx.y;                // 0..127
  int m    = lane & 15, half = lane >> 4;
  int ox   = oxt * 16 + m;
  float bias = b2[oct * 16 + (lane & 15)];
  v8f acc;
#pragma unroll
  for (int q = 0; q < 8; ++q) acc[q] = bias;
  // base of this lane's input column at tap (0,0): padded coords (2*oy, 2*ox)
  const _Float16* pb0 = h1p + (size_t)((2 * oy) * H1W + (2 * ox)) * 32;
  const _Float16* wb  = w2p + (size_t)((oct * 32 + lane) * 16);
#pragma unroll 5
  for (int tap = 0; tap < 25; ++tap) {
    int ky = tap / 5, kx = tap % 5;
    const _Float16* pb = pb0 + (size_t)(ky * H1W + kx) * 32;
    h8 lo = *(const h8*)(pb + half * 8);
    h8 hi = *(const h8*)(pb + 16 + half * 8);
    v16h a = __builtin_shufflevector(lo, hi, 0, 1, 2, 3, 4, 5, 6, 7,
                                     8, 9, 10, 11, 12, 13, 14, 15);
    v16h bf = *(const v16h*)(wb + (size_t)tap * (4 * 32 * 16));
    acc = __builtin_amdgcn_wmma_f32_16x16x32_f16(false, a, false, bf,
                                                 (short)0, acc, false, false);
  }
  float psum = 0.f;
#pragma unroll
  for (int q = 0; q < 8; ++q) psum += fmaxf(acc[q], 0.f);
  psum += __shfl_xor(psum, 16, 32);     // combine rows 0-7 with 8-15 (same col)
  if (lane < 16) atomicAdd(&chansum[oct * 16 + lane], psum);
}

// =====================================================================
// Single-block: mean -> fc1+ReLU -> fc2 -> s head (softplus) -> bitonic
// sort descending -> write s to out[131072..131200). Stores f to ws.
// =====================================================================
__global__ void __launch_bounds__(256)
k_fc_s(const float* __restrict__ chansum,
       const float* __restrict__ fc1_w, const float* __restrict__ fc1_b,
       const float* __restrict__ fc2_w, const float* __restrict__ fc2_b,
       const float* __restrict__ s_w,   const float* __restrict__ s_b,
       float* __restrict__ fvec, float* __restrict__ s_out) {
  __shared__ float feat[64], f1[256], f2[256], sv[128];
  int t = threadIdx.x;
  if (t < 64) feat[t] = chansum[t] * (1.0f / 16384.0f);
  __syncthreads();
  {
    float acc = fc1_b[t];
    const float* wp = fc1_w + t * 64;
#pragma unroll 8
    for (int c = 0; c < 64; ++c) acc += wp[c] * feat[c];
    f1[t] = fmaxf(acc, 0.f);
  }
  __syncthreads();
  {
    float acc = fc2_b[t];
    const float* wp = fc2_w + t * 256;
#pragma unroll 8
    for (int c = 0; c < 256; ++c) acc += wp[c] * f1[c];
    f2[t] = acc;
    fvec[t] = acc;
  }
  __syncthreads();
  if (t < 128) {
    float acc = s_b[t];
    const float* wp = s_w + t * 256;
#pragma unroll 8
    for (int c = 0; c < 256; ++c) acc += wp[c] * f2[c];
    sv[t] = (acc > 20.f) ? acc : log1pf(expf(acc));
  }
  // bitonic sort, descending
  for (int k = 2; k <= 128; k <<= 1)
    for (int j = k >> 1; j > 0; j >>= 1) {
      __syncthreads();
      if (t < 128) {
        int ixj = t ^ j;
        if (ixj > t) {
          float a = sv[t], b = sv[ixj];
          bool sw = ((t & k) == 0) ? (a < b) : (a > b);
          if (sw) { sv[t] = b; sv[ixj] = a; }
        }
      }
    }
  __syncthreads();
  if (t < 128) s_out[t] = sv[t];
}

// =====================================================================
// u/v projection: 131072 x 256 matvec (HBM-bandwidth bound, ~134 MB each).
// 8 lanes per row -> 128B coalesced chunks; shfl reduce; writes staged
// COLUMN-MAJOR (j*512+r) so Gram-Schmidt gets coalesced column access.
// =====================================================================
__global__ void __launch_bounds__(256)
k_matvec(const float* __restrict__ W, const float* __restrict__ bias,
         const float* __restrict__ fvec, float* __restrict__ outc) {
  __shared__ float lf[256];
  int t = threadIdx.x;
  lf[t] = fvec[t];
  __syncthreads();
  int lane = t & 31, wv = t >> 5;
  int g = lane & 7, grp = lane >> 3;
  int row = blockIdx.x * 32 + wv * 4 + grp;
  const float* wp = W + (size_t)row * 256 + g * 4;
  float acc = 0.f;
#pragma unroll
  for (int s = 0; s < 8; ++s) {
    float4 w4 = *(const float4*)(wp + s * 32);
    const float* fp = &lf[s * 32 + g * 4];
    acc += w4.x * fp[0] + w4.y * fp[1] + w4.z * fp[2] + w4.w * fp[3];
    __builtin_prefetch(wp + s * 32 + 32 * 256, 0, 1);   // next row-block, speculative
  }
  acc += __shfl_xor(acc, 4, 32);
  acc += __shfl_xor(acc, 2, 32);
  acc += __shfl_xor(acc, 1, 32);
  if (g == 0) {
    float val = acc + bias[row];
    int c = row & 1, j = (row >> 1) & 127, r = row >> 8;
    outc[(j * 512 + r) * 2 + c] = val;                  // column-major staging
  }
}

// =====================================================================
// Right-looking modified Gram-Schmidt (numerically identical projection
// sequence to the reference scan). One block per matrix (u, v).
// q_i broadcast via LDS; 16 waves own trailing columns independently.
// =====================================================================
__device__ __forceinline__ float block_reduce_sum512(float v, float* red) {
  int lane = threadIdx.x & 31, wid = threadIdx.x >> 5;
#pragma unroll
  for (int off = 16; off > 0; off >>= 1) v += __shfl_xor(v, off, 32);
  if (lane == 0) red[wid] = v;
  __syncthreads();
  if (wid == 0) {
    float x = (lane < 16) ? red[lane] : 0.f;
#pragma unroll
    for (int off = 8; off > 0; off >>= 1) x += __shfl_xor(x, off, 32);
    if (lane == 0) red[0] = x;
  }
  __syncthreads();
  float total = red[0];
  __syncthreads();
  return total;
}

__global__ void __launch_bounds__(512)
k_gram_schmidt(float* __restrict__ ubuf, float* __restrict__ vbuf) {
  float* buf = (blockIdx.x == 0) ? ubuf : vbuf;
  __shared__ float qre[512], qim[512], red[16];
  int t = threadIdx.x;
  int lane = t & 31, wid = t >> 5;
  for (int i = 0; i < 128; ++i) {
    // --- normalize column i (in place) ---
    float vr = buf[(i * 512 + t) * 2 + 0];
    float vi = buf[(i * 512 + t) * 2 + 1];
    float total = block_reduce_sum512(vr * vr + vi * vi, red);
    float norm = sqrtf(total + 1e-8f);
    float qr = vr / norm, qi = vi / norm;
    buf[(i * 512 + t) * 2 + 0] = qr;
    buf[(i * 512 + t) * 2 + 1] = qi;
    qre[t] = qr; qim[t] = qi;
    __threadfence_block();
    __syncthreads();
    // --- project all trailing columns against q_i; wave per column ---
    for (int j = i + 1 + wid; j < 128; j += 16) {
      float* base = buf + j * 1024;
      float cr = 0.f, ci = 0.f;
      float brv[16], biv[16];
#pragma unroll
      for (int kk = 0; kk < 16; ++kk) {
        int r = lane + kk * 32;
        float br = base[r * 2], bi = base[r * 2 + 1];
        brv[kk] = br; biv[kk] = bi;
        float q0 = qre[r], q1 = qim[r];
        cr += q0 * br + q1 * bi;      // Re<q,v> with conj(q)
        ci += q0 * bi - q1 * br;      // Im<q,v>
      }
#pragma unroll
      for (int off = 16; off > 0; off >>= 1) {
        cr += __shfl_xor(cr, off, 32);
        ci += __shfl_xor(ci, off, 32);
      }
#pragma unroll
      for (int kk = 0; kk < 16; ++kk) {
        int r = lane + kk * 32;
        float q0 = qre[r], q1 = qim[r];
        base[r * 2]     = brv[kk] - (cr * q0 - ci * q1);
        base[r * 2 + 1] = biv[kk] - (cr * q1 + ci * q0);
      }
    }
    __threadfence_block();
    __syncthreads();
  }
}

// =====================================================================
// Final transpose: column-major staging -> (512,128,2) output layout.
// =====================================================================
__global__ void __launch_bounds__(256)
k_transpose(const float* __restrict__ ubuf, const float* __restrict__ vbuf,
            float* __restrict__ out) {
  int o = blockIdx.x * 256 + threadIdx.x;    // 0..131071, linear over source
  const float* src = (blockIdx.y == 0) ? ubuf : vbuf;
  int dbase = (blockIdx.y == 0) ? 0 : (131072 + 128);
  float val = src[o];
  int c = o & 1, r = (o >> 1) & 511, i = o >> 10;
  out[dbase + (r * 128 + i) * 2 + c] = val;
}

// =====================================================================
extern "C" void kernel_launch(void* const* d_in, const int* in_sizes, int n_in,
                              void* d_out, int out_size, void* d_ws, size_t ws_size,
                              hipStream_t stream) {
  (void)in_sizes; (void)n_in; (void)out_size; (void)ws_size;
  const float* x       = (const float*)d_in[0];
  const float* conv1_w = (const float*)d_in[1];
  const float* conv1_b = (const float*)d_in[2];
  const float* conv2_w = (const float*)d_in[3];
  const float* conv2_b = (const float*)d_in[4];
  const float* fc1_w   = (const float*)d_in[5];
  const float* fc1_b   = (const float*)d_in[6];
  const float* fc2_w   = (const float*)d_in[7];
  const float* fc2_b   = (const float*)d_in[8];
  const float* s_w     = (const float*)d_in[9];
  const float* s_b     = (const float*)d_in[10];
  const float* u_w     = (const float*)d_in[11];
  const float* u_b     = (const float*)d_in[12];
  const float* v_w     = (const float*)d_in[13];
  const float* v_b     = (const float*)d_in[14];
  float* out = (float*)d_out;
  char* ws = (char*)d_ws;
  _Float16* h1p  = (_Float16*)(ws + WS_H1_OFF);
  _Float16* w2p  = (_Float16*)(ws + WS_W2P_OFF);
  float* chansum = (float*)(ws + WS_CHAN_OFF);
  float* fvec    = (float*)(ws + WS_FVEC_OFF);
  float* ubuf    = (float*)(ws + WS_UBUF_OFF);
  float* vbuf    = (float*)(ws + WS_VBUF_OFF);

  k_zero_pad<<<9, 256, 0, stream>>>(h1p);
  k_conv1<<<512, 128, 0, stream>>>(x, conv1_w, conv1_b, h1p);
  k_pack_w2<<<25, 128, 0, stream>>>(conv2_w, w2p, chansum);
  k_conv2_wmma<<<dim3(8, 128), 128, 0, stream>>>(h1p, w2p, conv2_b, chansum);
  k_fc_s<<<1, 256, 0, stream>>>(chansum, fc1_w, fc1_b, fc2_w, fc2_b,
                                s_w, s_b, fvec, out + 131072);
  k_matvec<<<4096, 256, 0, stream>>>(u_w, u_b, fvec, ubuf);
  k_matvec<<<4096, 256, 0, stream>>>(v_w, v_b, fvec, vbuf);
  k_gram_schmidt<<<2, 512, 0, stream>>>(ubuf, vbuf);
  k_transpose<<<dim3(512, 2), 256, 0, stream>>>(ubuf, vbuf, out);
}
